// LSTMClassifier_11639361372736
// MI455X (gfx1250) — matrix operand order repaired and checked by reading, not compiled, not run on hardware
//
#include <hip/hip_runtime.h>

#define B_ 512
#define T_ 512
#define D_ 256
#define H_ 512

typedef __attribute__((ext_vector_type(16))) _Float16 v16h;
typedef __attribute__((ext_vector_type(8)))  float    v8f;

#define WMMA_F16(A, Bf, C) \
    __builtin_amdgcn_wmma_f32_16x16x32_f16(false, (A), false, (Bf), (short)0, (C), false, false)

// ---------- WMMA fragment loaders (CDNA5 ISA layouts, wave32) ----------
// A fragment: 16(M) x 32(K). Lanes 0-15: rows M=lane, K = {0..7, 16..23};
// lanes 16-31: rows M=lane-16, K = {8..15, 24..31}. (ISA 7.12.2)
__device__ __forceinline__ v16h load_a_f32(const float* __restrict__ src, long lda, int lane) {
    const int   row = lane & 15;
    const int   kb  = (lane >> 4) << 3;       // 0 or 8
    const float* p  = src + (long)row * lda + kb;
    v16h a;
#pragma unroll
    for (int e = 0; e < 8; ++e) a[e]     = (_Float16)p[e];
#pragma unroll
    for (int e = 0; e < 8; ++e) a[e + 8] = (_Float16)p[16 + e];
    return a;
}

__device__ __forceinline__ v16h load_a_f16(const _Float16* __restrict__ src, long lda, int lane) {
    const int row = lane & 15;
    const int kb  = (lane >> 4) << 3;
    const _Float16* p = src + (long)row * lda + kb;
    v16h a;
#pragma unroll
    for (int e = 0; e < 8; ++e) a[e]     = p[e];
#pragma unroll
    for (int e = 0; e < 8; ++e) a[e + 8] = p[16 + e];
    return a;
}

// B fragment: 32(K) x 16(N). Lanes 0-15: N=lane, K = {0..7, 16..23}; lanes 16-31: N=lane-16,
// K = {8..15, 24..31}. Row-major W[n][k] (i.e. B^T), same K order as the A fragment.
__device__ __forceinline__ v16h load_b_w(const _Float16* __restrict__ w, long ldw, int lane) {
    const int ncol = lane & 15;
    const int kb   = (lane >> 4) << 3;        // 0 or 8
    const _Float16* p = w + (long)ncol * ldw + kb;
    v16h b;
#pragma unroll
    for (int e = 0; e < 8; ++e) b[e]     = p[e];
#pragma unroll
    for (int e = 0; e < 8; ++e) b[e + 8] = p[16 + e];
    return b;
}

__device__ __forceinline__ float sigmoidf_(float x) {
    return 1.0f / (1.0f + __expf(-x));
}

// ---------- Kernel 1: prep (f32->f16 weights, combined bias, zero state) ----------
__global__ __launch_bounds__(256) void lstm_prep_kernel(
    const float* __restrict__ Wih, const float* __restrict__ Whh,
    const float* __restrict__ bih, const float* __restrict__ bhh,
    _Float16* __restrict__ WihH, _Float16* __restrict__ WhhH,
    float* __restrict__ bias, float* __restrict__ c, _Float16* __restrict__ h0)
{
    const long i = (long)blockIdx.x * blockDim.x + threadIdx.x;
    if (i < (long)4 * H_ * D_) WihH[i] = (_Float16)Wih[i];
    if (i < (long)4 * H_ * H_) WhhH[i] = (_Float16)Whh[i];
    if (i < 4 * H_)            bias[i] = bih[i] + bhh[i];
    if (i < (long)B_ * H_) { c[i] = 0.0f; h0[i] = (_Float16)0.0f; }
}

// ---------- Kernel 2: one LSTM time step, fully fused, software-pipelined ----------
// gates = x[:,t,:] @ Wih^T + h @ Whh^T + bias ; c,h update in-register.
// One wave per 16x16 (batch x hidden) tile, 4 gate accumulators per wave.
__global__ __launch_bounds__(128) void lstm_step_kernel(
    const float* __restrict__ x, int t,
    const _Float16* __restrict__ Wih,   // [4H, D] f16
    const _Float16* __restrict__ Whh,   // [4H, H] f16
    const float* __restrict__ bias,     // [4H] = b_ih + b_hh
    float* __restrict__ c,              // [B, H]
    const _Float16* __restrict__ hin,   // [B, H] f16 (read)
    _Float16* __restrict__ hout)        // [B, H] f16 (write)
{
    const int lane = threadIdx.x & 31;
    const int wave = blockIdx.x * (blockDim.x >> 5) + (threadIdx.x >> 5);
    const int tilesN = H_ / 16;                 // 32
    const int m0 = (wave / tilesN) * 16;        // batch tile
    const int n0 = (wave % tilesN) * 16;        // hidden tile

    const long ldx = (long)T_ * D_;
    const float*    xbase = x   + (long)m0 * ldx + (long)t * D_;
    const _Float16* hbase = hin + (long)m0 * H_;
    const _Float16* wih0 = Wih + (long)(0 * H_ + n0) * D_;
    const _Float16* wih1 = Wih + (long)(1 * H_ + n0) * D_;
    const _Float16* wih2 = Wih + (long)(2 * H_ + n0) * D_;
    const _Float16* wih3 = Wih + (long)(3 * H_ + n0) * D_;
    const _Float16* whh0 = Whh + (long)(0 * H_ + n0) * H_;
    const _Float16* whh1 = Whh + (long)(1 * H_ + n0) * H_;
    const _Float16* whh2 = Whh + (long)(2 * H_ + n0) * H_;
    const _Float16* whh3 = Whh + (long)(3 * H_ + n0) * H_;

    // ---- hoist latency-critical tail operands out of the serial GEMM path ----
    const int n  = lane & 15;
    const int mh = (lane >> 4) << 3;
    const float bI = bias[0 * H_ + n0 + n];
    const float bF = bias[1 * H_ + n0 + n];
    const float bG = bias[2 * H_ + n0 + n];
    const float bO = bias[3 * H_ + n0 + n];
    long  oidx[8];
    float cv[8];
#pragma unroll
    for (int r = 0; r < 8; ++r) {
        oidx[r] = (long)(m0 + mh + r) * H_ + n0 + n;
        cv[r]   = c[oidx[r]];
    }

    // ---- warm L2 for the NEXT timestep's x slice (global_prefetch_b8) ----
    if (t + 1 < T_) {
        const char* pf = (const char*)(xbase + D_ + (long)(lane & 15) * ldx) +
                         ((lane >> 4) << 9);   // two 512B halves of each 1KB row
#pragma unroll
        for (int i = 0; i < 4; ++i) __builtin_prefetch(pf + i * 128, 0, 1);
    }

    v8f acc0 = {}, acc1 = {}, acc2 = {}, acc3 = {};

    // ---- software-pipelined K loop: preload chunk 0 ----
    v16h aC  = load_a_f32(xbase, ldx, lane);
    v16h b0C = load_b_w(wih0, D_, lane);
    v16h b1C = load_b_w(wih1, D_, lane);
    v16h b2C = load_b_w(wih2, D_, lane);
    v16h b3C = load_b_w(wih3, D_, lane);

    // Phase 1: x[:,t,:] @ Wih^T  (K = D = 256), prefetch k+32 before WMMA of k
#pragma unroll
    for (int k0 = 0; k0 < D_ - 32; k0 += 32) {
        const int k1 = k0 + 32;
        v16h aN  = load_a_f32(xbase + k1, ldx, lane);
        v16h b0N = load_b_w(wih0 + k1, D_, lane);
        v16h b1N = load_b_w(wih1 + k1, D_, lane);
        v16h b2N = load_b_w(wih2 + k1, D_, lane);
        v16h b3N = load_b_w(wih3 + k1, D_, lane);
        acc0 = WMMA_F16(aC, b0C, acc0);
        acc1 = WMMA_F16(aC, b1C, acc1);
        acc2 = WMMA_F16(aC, b2C, acc2);
        acc3 = WMMA_F16(aC, b3C, acc3);
        aC = aN; b0C = b0N; b1C = b1N; b2C = b2N; b3C = b3N;
    }

    // Bridge: prefetch first recurrent chunk, finish last x chunk
    {
        v16h aN  = load_a_f16(hbase, H_, lane);
        v16h b0N = load_b_w(whh0, H_, lane);
        v16h b1N = load_b_w(whh1, H_, lane);
        v16h b2N = load_b_w(whh2, H_, lane);
        v16h b3N = load_b_w(whh3, H_, lane);
        acc0 = WMMA_F16(aC, b0C, acc0);
        acc1 = WMMA_F16(aC, b1C, acc1);
        acc2 = WMMA_F16(aC, b2C, acc2);
        acc3 = WMMA_F16(aC, b3C, acc3);
        aC = aN; b0C = b0N; b1C = b1N; b2C = b2N; b3C = b3N;
    }

    // Phase 2: h @ Whh^T  (K = H = 512)
#pragma unroll 5
    for (int k0 = 0; k0 < H_ - 32; k0 += 32) {
        const int k1 = k0 + 32;
        v16h aN  = load_a_f16(hbase + k1, H_, lane);
        v16h b0N = load_b_w(whh0 + k1, H_, lane);
        v16h b1N = load_b_w(whh1 + k1, H_, lane);
        v16h b2N = load_b_w(whh2 + k1, H_, lane);
        v16h b3N = load_b_w(whh3 + k1, H_, lane);
        acc0 = WMMA_F16(aC, b0C, acc0);
        acc1 = WMMA_F16(aC, b1C, acc1);
        acc2 = WMMA_F16(aC, b2C, acc2);
        acc3 = WMMA_F16(aC, b3C, acc3);
        aC = aN; b0C = b0N; b1C = b1N; b2C = b2N; b3C = b3N;
    }
    // Epilogue chunk
    acc0 = WMMA_F16(aC, b0C, acc0);
    acc1 = WMMA_F16(aC, b1C, acc1);
    acc2 = WMMA_F16(aC, b2C, acc2);
    acc3 = WMMA_F16(aC, b3C, acc3);

    // ---- gate nonlinearity + state update (C/D layout: lane half -> M offset 0/8) ----
#pragma unroll
    for (int r = 0; r < 8; ++r) {
        const float gi = sigmoidf_(acc0[r] + bI);
        const float gf = sigmoidf_(acc1[r] + bF);
        const float gg = tanhf(acc2[r] + bG);
        const float go = sigmoidf_(acc3[r] + bO);
        float cn = gf * cv[r] + gi * gg;
        c[oidx[r]] = cn;
        hout[oidx[r]] = (_Float16)(go * tanhf(cn));
    }
}

// ---------- Kernel 3: FC head: sigmoid(h @ W_fc^T + b_fc) ----------
__global__ __launch_bounds__(256) void lstm_fc_kernel(
    const _Float16* __restrict__ h, const float* __restrict__ Wfc,
    const float* __restrict__ bfc, float* __restrict__ out)
{
    const int b = blockIdx.x * blockDim.x + threadIdx.x;
    if (b < B_) {
        float s = 0.0f;
#pragma unroll 8
        for (int k = 0; k < H_; ++k) s += (float)h[(long)b * H_ + k] * Wfc[k];
        out[b] = sigmoidf_(s + bfc[0]);
    }
}

extern "C" void kernel_launch(void* const* d_in, const int* in_sizes, int n_in,
                              void* d_out, int out_size, void* d_ws, size_t ws_size,
                              hipStream_t stream) {
    (void)in_sizes; (void)n_in; (void)out_size; (void)ws_size;
    const float* x   = (const float*)d_in[0];
    const float* Wih = (const float*)d_in[1];
    const float* Whh = (const float*)d_in[2];
    const float* bih = (const float*)d_in[3];
    const float* bhh = (const float*)d_in[4];
    const float* Wfc = (const float*)d_in[5];
    const float* bfc = (const float*)d_in[6];
    float* out = (float*)d_out;

    // Workspace layout (total ~5.26 MB)
    char* ws = (char*)d_ws;
    _Float16* WihH = (_Float16*)(ws + 0);              // 4H*D f16  = 1,048,576 B
    _Float16* WhhH = (_Float16*)(ws + 1048576);        // 4H*H f16  = 2,097,152 B
    float*    bias = (float*)   (ws + 3145728);        // 4H  f32   =     8,192 B
    float*    c    = (float*)   (ws + 3153920);        // B*H f32   = 1,048,576 B
    _Float16* hb0  = (_Float16*)(ws + 4202496);        // B*H f16   =   524,288 B
    _Float16* hb1  = (_Float16*)(ws + 4726784);        // B*H f16   =   524,288 B

    // Prep: convert weights, fuse biases, zero h/c. 4H*H = 1,048,576 threads.
    lstm_prep_kernel<<<4096, 256, 0, stream>>>(Wih, Whh, bih, bhh, WihH, WhhH, bias, c, hb0);

    // Recurrence: T stream-ordered launches; h double-buffered.
    _Float16* hbuf[2] = { hb0, hb1 };
    for (int t = 0; t < T_; ++t) {
        lstm_step_kernel<<<256, 128, 0, stream>>>(x, t, WihH, WhhH, bias, c,
                                                  hbuf[t & 1], hbuf[(t + 1) & 1]);
    }
    // After T=512 steps final h is in hbuf[0].
    lstm_fc_kernel<<<2, 256, 0, stream>>>(hbuf[0], Wfc, bfc, out);
}
